// FusedTriangleMultiplicationOpti_24292335027055
// MI455X (gfx1250) — compile-verified, matrix operand order checked by reading, not run on hardware
//
#include <hip/hip_runtime.h>

#define N_ 512
#define NM (N_*N_)      // 262144 grid positions
#define CZ 128
#define CI 128

typedef __attribute__((ext_vector_type(16))) __bf16 v16bf;
typedef __attribute__((ext_vector_type(4)))  __bf16 v4bf;
typedef __attribute__((ext_vector_type(8)))  float  v8f;
typedef __attribute__((ext_vector_type(4)))  float  v4f;
typedef __attribute__((ext_vector_type(4)))  int    v4i;

#if defined(__has_builtin)
#  if __has_builtin(__builtin_amdgcn_global_load_async_to_lds_b128)
#    define HAVE_ASYNC_BUILTIN 1
#  else
#    define HAVE_ASYNC_BUILTIN 0
#  endif
#  if __has_builtin(__builtin_amdgcn_s_wait_asynccnt)
#    define HAVE_WAITASYNC_BUILTIN 1
#  else
#    define HAVE_WAITASYNC_BUILTIN 0
#  endif
#else
#  define HAVE_ASYNC_BUILTIN 0
#  define HAVE_WAITASYNC_BUILTIN 0
#endif

__device__ __forceinline__ float sigmoidf_(float x){ return 1.0f/(1.0f+__expf(-x)); }

__device__ __forceinline__ v8f zero8(){ v8f z; for (int i=0;i<8;++i) z[i]=0.0f; return z; }

__device__ __forceinline__ v8f wmma_bf16(v16bf a, v16bf b, v8f c){
  // D = A(16x32 bf16) * B(32x16 bf16) + C(16x16 f32)
  return __builtin_amdgcn_wmma_f32_16x16x32_bf16(false, a, false, b, (short)0, c, false, false);
}

// Per-lane fragment load for 16-bit A/B operands (16 rows x 32 k), per ISA layout:
// lane r=lane&15 row, hi half selects K {0..7,16..23} vs {8..15,24..31}.
__device__ __forceinline__ v16bf load_frag(const __bf16* base, int strideElems, int lane){
  const int r  = lane & 15;
  const int hi = (lane >> 4) & 1;
  const char* p = (const char*)(base + r*strideElems);
  v4f c0 = *(const v4f*)(p + hi*16);
  v4f c1 = *(const v4f*)(p + 32 + hi*16);
  v16bf out;
  __builtin_memcpy(&out, &c0, 16);
  __builtin_memcpy((char*)&out + 16, &c1, 16);
  return out;
}

// Async global -> LDS copy of 16 bytes (per enabled lane), tracked by ASYNCcnt.
__device__ __forceinline__ void async_b128(const void* g, void* l){
#if HAVE_ASYNC_BUILTIN
  typedef __attribute__((address_space(1))) v4i as1_v4i;
  typedef __attribute__((address_space(3))) v4i as3_v4i;
  __builtin_amdgcn_global_load_async_to_lds_b128(
      (as1_v4i*)(unsigned long long)(__UINTPTR_TYPE__)g,
      (as3_v4i*)(unsigned int)(__UINTPTR_TYPE__)l,
      0, 0);
#else
  asm volatile("global_load_async_to_lds_b128 %0, %1, off"
               :: "v"((unsigned int)(__UINTPTR_TYPE__)l),
                  "v"((unsigned long long)(__UINTPTR_TYPE__)g)
               : "memory");
#endif
}

#if HAVE_WAITASYNC_BUILTIN
#define WAIT_ASYNC(n) __builtin_amdgcn_s_wait_asynccnt(n)
#else
#define WAIT_ASYNC(n) asm volatile("s_wait_asynccnt %0" :: "i"(n) : "memory")
#endif

// ---------------- k0: cast weights to bf16 (Wcat = [proj_w;gate_w;gl_w], Wout = outp_w)
__global__ __launch_bounds__(256) void k0_cast_weights(
    const float* __restrict__ proj_w, const float* __restrict__ gate_w,
    const float* __restrict__ gl_w,   const float* __restrict__ outp_w,
    __bf16* __restrict__ Wcat, __bf16* __restrict__ Wout)
{
  int i = blockIdx.x*256 + threadIdx.x;
  if (i < 32768)       Wcat[i]       = (__bf16)proj_w[i];          // rows 0..255
  else if (i < 65536)  Wcat[i]       = (__bf16)gate_w[i-32768];    // rows 256..511
  else if (i < 81920)  Wcat[i]       = (__bf16)gl_w[i-65536];      // rows 512..639
  else if (i < 98304)  Wout[i-81920] = (__bf16)outp_w[i-81920];
}

// ---------------- k1: LayerNorm(act) -> bf16, one wave32 per grid row
__global__ __launch_bounds__(256) void k1_ln_cast(
    const float* __restrict__ act, const float* __restrict__ w,
    const float* __restrict__ b, __bf16* __restrict__ left)
{
  const int lane = threadIdx.x & 31;
  const int wv   = threadIdx.x >> 5;
  const size_t row = (size_t)blockIdx.x*8 + wv;
  v4f v = *(const v4f*)(act + row*CZ + lane*4);
  float s = v[0]+v[1]+v[2]+v[3];
  float q = v[0]*v[0]+v[1]*v[1]+v[2]*v[2]+v[3]*v[3];
  for (int off=16; off>0; off>>=1){ s += __shfl_xor(s, off, 32); q += __shfl_xor(q, off, 32); }
  float mean = s * (1.0f/CZ);
  float rstd = rsqrtf(q*(1.0f/CZ) - mean*mean + 1e-5f);
  v4bf o;
  for (int u=0;u<4;++u){
    int c = lane*4+u;
    o[u] = (__bf16)((v[u]-mean)*rstd*w[c] + b[c]);
  }
  *(v4bf*)(left + row*CZ + lane*4) = o;
}

// ---------------- k2: fused proj/gate/gl GEMM + gating epilogue, channel-major bf16 outputs
__global__ __launch_bounds__(256) void k2_proj_gate(
    const __bf16* __restrict__ left, const __bf16* __restrict__ Wcat,
    const float* __restrict__ mask,
    const float* __restrict__ proj_b, const float* __restrict__ gate_b,
    const float* __restrict__ gl_b,
    __bf16* __restrict__ leftp, __bf16* __restrict__ rightp,
    float* __restrict__ gl_gate)
{
  __shared__ __bf16 At[64*40];   // 64 rows x 32 k, stride 40 (80B, bank-spread)
  __shared__ __bf16 Tr[64*72];   // transpose staging: [ncol][m], stride 72 (144B)
  const int tid = threadIdx.x, lane = tid & 31, wv = tid >> 5;
  const int m0 = blockIdx.x*64;
  const int cg = blockIdx.y;              // column group of 64 within 256 proj cols
  const int mloc = (wv & 3)*16;
  const int nh   = (wv >> 2)*32;
  const bool has_gl = (cg < 2);

  v8f ap0=zero8(), ap1=zero8(), ag0=zero8(), ag1=zero8(), al0=zero8(), al1=zero8();

  for (int k0=0; k0<CZ; k0+=32){
    const int r = tid >> 2, ch = tid & 3;
    v4f d = *(const v4f*)(left + (size_t)(m0+r)*CZ + k0 + ch*8);
    __syncthreads();
    *(v4f*)(At + r*40 + ch*8) = d;
    __syncthreads();
    v16bf a   = load_frag(At + mloc*40, 40, lane);
    v16bf bp0 = load_frag(Wcat + (size_t)(      cg*64 + nh +  0)*CZ + k0, CZ, lane);
    v16bf bp1 = load_frag(Wcat + (size_t)(      cg*64 + nh + 16)*CZ + k0, CZ, lane);
    v16bf bg0 = load_frag(Wcat + (size_t)(256 + cg*64 + nh +  0)*CZ + k0, CZ, lane);
    v16bf bg1 = load_frag(Wcat + (size_t)(256 + cg*64 + nh + 16)*CZ + k0, CZ, lane);
    ap0 = wmma_bf16(a, bp0, ap0);
    ap1 = wmma_bf16(a, bp1, ap1);
    ag0 = wmma_bf16(a, bg0, ag0);
    ag1 = wmma_bf16(a, bg1, ag1);
    if (has_gl){
      v16bf bl0 = load_frag(Wcat + (size_t)(512 + cg*64 + nh +  0)*CZ + k0, CZ, lane);
      v16bf bl1 = load_frag(Wcat + (size_t)(512 + cg*64 + nh + 16)*CZ + k0, CZ, lane);
      al0 = wmma_bf16(a, bl0, al0);
      al1 = wmma_bf16(a, bl1, al1);
    }
  }

  const int hi = (lane>>4)&1, nl = lane & 15;
  for (int t=0;t<2;++t){
    v8f P = t ? ap1 : ap0;
    v8f G = t ? ag1 : ag0;
    int ncol = nh + t*16 + nl;
    int c    = cg*64 + ncol;            // 0..255
    float pb = proj_b[c], gb = gate_b[c];
    for (int e=0;e<8;++e){
      int ml = mloc + e + hi*8;
      int m  = m0 + ml;
      float p = mask[m] * (P[e] + pb) * sigmoidf_(G[e] + gb);
      Tr[ncol*72 + ml] = (__bf16)p;
    }
  }
  if (has_gl){
    for (int t=0;t<2;++t){
      v8f L = t ? al1 : al0;
      int ncol = nh + t*16 + nl;
      int c = cg*64 + ncol;             // 0..127
      float lb = gl_b[c];
      for (int e=0;e<8;++e){
        int m = m0 + mloc + e + hi*8;
        gl_gate[(size_t)m*CI + c] = sigmoidf_(L[e] + lb);
      }
    }
  }
  __syncthreads();
  {
    // cooperative transposed writeout: contiguous 32B runs along m per channel
    int ncol = tid >> 2, qq = tid & 3;
    int c = cg*64 + ncol;
    v4f t0 = *(const v4f*)(Tr + ncol*72 + qq*16);
    v4f t1 = *(const v4f*)(Tr + ncol*72 + qq*16 + 8);
    __bf16* dst = (c < CI) ? (leftp  + (size_t)c     *NM + m0 + qq*16)
                           : (rightp + (size_t)(c-CI)*NM + m0 + qq*16);
    *(v4f*)dst     = t0;
    *(v4f*)(dst+8) = t1;
  }
}

// ---------------- k3: 128 batched 512x512x512 NT GEMMs (L2-resident operands).
// 128 threads / 4 waves; each wave owns a 32x32 quadrant (4 WMMAs : 4 frag loads).
// Double-buffered LDS fed by async global->LDS b128 copies tracked with ASYNCcnt.
__global__ __launch_bounds__(128) void k3_einsum(
    const __bf16* __restrict__ leftp, const __bf16* __restrict__ rightp,
    float* __restrict__ out_pre)
{
  __shared__ __bf16 At[2][64*40];
  __shared__ __bf16 Bt[2][64*40];
  const int tid = threadIdx.x, lane = tid & 31, wv = tid >> 5;
  const int i0 = blockIdx.x*64, j0 = blockIdx.y*64, c = blockIdx.z;
  const __bf16* A = leftp  + (size_t)c*NM;
  const __bf16* B = rightp + (size_t)c*NM;
  const int mh = (wv & 1)*32, nh = (wv >> 1)*32;

  // staging: threads 0..63 stream A rows, 64..127 stream B rows (one 64B row each)
  const int srow = tid & 63;
  const __bf16* gbase = (tid < 64) ? (A + (size_t)(i0+srow)*N_)
                                   : (B + (size_t)(j0+srow)*N_);

  v8f acc00=zero8(), acc01=zero8(), acc10=zero8(), acc11=zero8();

  // issue tile kt (32 k-values) into buffer kt&1: 4 async b128 per thread
  auto stage = [&](int kt){
    const int buf = kt & 1;
    const __bf16* g = gbase + kt*32;
    __bf16* l = ((tid < 64) ? At[buf] : Bt[buf]) + srow*40;
    async_b128(g +  0, l +  0);
    async_b128(g +  8, l +  8);
    async_b128(g + 16, l + 16);
    async_b128(g + 24, l + 24);
  };

  stage(0);
  for (int kt = 0; kt < 16; ++kt){
    if (kt < 15){
      stage(kt + 1);
      WAIT_ASYNC(4);   // next tile's 4 still in flight => current tile landed
    } else {
      WAIT_ASYNC(0);
    }
    __syncthreads();   // all waves' async copies visible
    const int buf = kt & 1;
    v16bf a0 = load_frag(At[buf] + (mh   )*40, 40, lane);
    v16bf a1 = load_frag(At[buf] + (mh+16)*40, 40, lane);
    v16bf b0 = load_frag(Bt[buf] + (nh   )*40, 40, lane);
    v16bf b1 = load_frag(Bt[buf] + (nh+16)*40, 40, lane);
    acc00 = wmma_bf16(a0, b0, acc00);
    acc01 = wmma_bf16(a0, b1, acc01);
    acc10 = wmma_bf16(a1, b0, acc10);
    acc11 = wmma_bf16(a1, b1, acc11);
    __syncthreads();   // everyone done reading this buffer before it is refilled
  }

  const int hi = (lane>>4)&1, nl = lane & 15;
  float* dst = out_pre + (size_t)c*NM;
  auto store_tile = [&](v8f Ac, int mt, int nt){
    int n = j0 + nt + nl;
    for (int e=0;e<8;++e){
      int i = i0 + mt + e + hi*8;
      dst[(size_t)i*N_ + n] = Ac[e];
    }
  };
  store_tile(acc00, mh,    nh   );
  store_tile(acc01, mh,    nh+16);
  store_tile(acc10, mh+16, nh   );
  store_tile(acc11, mh+16, nh+16);
}

// ---------------- k4a: per-(i,j) mean / rstd over channels (channel-major reads coalesce over m)
__global__ __launch_bounds__(256) void k4a_stats(
    const float* __restrict__ out_pre, float* __restrict__ meanA, float* __restrict__ rstdA)
{
  int m = blockIdx.x*256 + threadIdx.x;
  float s=0.f, q=0.f;
  for (int c=0;c<CI;++c){ float v = out_pre[(size_t)c*NM + m]; s+=v; q+=v*v; }
  float mean = s*(1.0f/CI);
  meanA[m] = mean;
  rstdA[m] = rsqrtf(q*(1.0f/CI) - mean*mean + 1e-5f);
}

// ---------------- k4b: LN-apply + output projection (WMMA) + gl gating -> d_out
__global__ __launch_bounds__(256) void k4b_outproj(
    const float* __restrict__ out_pre,
    const float* __restrict__ meanA, const float* __restrict__ rstdA,
    const float* __restrict__ cn_w, const float* __restrict__ cn_b,
    const __bf16* __restrict__ Wout, const float* __restrict__ outp_b,
    const float* __restrict__ gl_gate, float* __restrict__ out)
{
  __shared__ __bf16 At[64*136];   // 64 m x 128 c, stride 136 (272B, 16B-aligned, bank-spread)
  __shared__ __bf16 Bw[128*136];  // outp_w staged
  __shared__ float  smean[64];
  __shared__ float  srstd[64];
  const int tid = threadIdx.x, lane = tid & 31, wv = tid >> 5;
  const int m0 = blockIdx.x*64;

  if (tid < 64){ smean[tid] = meanA[m0+tid]; srstd[tid] = rstdA[m0+tid]; }
  for (int it = tid; it < 2048; it += 256){
    int rr = it >> 4, ch = it & 15;
    *(v4f*)(Bw + rr*136 + ch*8) = *(const v4f*)(Wout + rr*128 + ch*8);
  }
  __syncthreads();

  // Transpose channel-major f32 -> row-major normalized bf16 A tile
  for (int it = tid; it < 512; it += 256){
    int cc = it >> 2, ch = it & 3;
    float wn = cn_w[cc], bn = cn_b[cc];
    const float* src = out_pre + (size_t)cc*NM + m0 + ch*16;
    for (int u=0; u<16; u+=4){
      v4f v = *(const v4f*)(src + u);
      int mm = ch*16 + u;
      At[(mm+0)*136 + cc] = (__bf16)((v[0]-smean[mm+0])*srstd[mm+0]*wn + bn);
      At[(mm+1)*136 + cc] = (__bf16)((v[1]-smean[mm+1])*srstd[mm+1]*wn + bn);
      At[(mm+2)*136 + cc] = (__bf16)((v[2]-smean[mm+2])*srstd[mm+2]*wn + bn);
      At[(mm+3)*136 + cc] = (__bf16)((v[3]-smean[mm+3])*srstd[mm+3]*wn + bn);
    }
  }
  __syncthreads();

  const int mloc = (wv & 3)*16, nh = (wv >> 2)*64;
  v8f acc[4]; for (int t=0;t<4;++t) acc[t] = zero8();
  for (int k0=0; k0<CI; k0+=32){
    v16bf a = load_frag(At + mloc*136 + k0, 136, lane);
    for (int t=0;t<4;++t){
      v16bf bw = load_frag(Bw + (nh + t*16)*136 + k0, 136, lane);
      acc[t] = wmma_bf16(a, bw, acc[t]);
    }
  }

  const int hi = (lane>>4)&1, nl = lane & 15;
  for (int t=0;t<4;++t){
    int n = nh + t*16 + nl;
    float ob = outp_b[n];
    for (int e=0;e<8;++e){
      int m = m0 + mloc + e + hi*8;
      out[(size_t)m*CI + n] = (acc[t][e] + ob) * gl_gate[(size_t)m*CI + n];
    }
  }
}

extern "C" void kernel_launch(void* const* d_in, const int* in_sizes, int n_in,
                              void* d_out, int out_size, void* d_ws, size_t ws_size,
                              hipStream_t stream)
{
  (void)in_sizes; (void)n_in; (void)out_size; (void)ws_size;
  const float* act    = (const float*)d_in[0];
  const float* mask   = (const float*)d_in[1];
  const float* ln_w   = (const float*)d_in[2];
  const float* ln_b   = (const float*)d_in[3];
  const float* proj_w = (const float*)d_in[4];
  const float* proj_b = (const float*)d_in[5];
  const float* gate_w = (const float*)d_in[6];
  const float* gate_b = (const float*)d_in[7];
  const float* cn_w   = (const float*)d_in[8];
  const float* cn_b   = (const float*)d_in[9];
  const float* outp_w = (const float*)d_in[10];
  const float* outp_b = (const float*)d_in[11];
  const float* gl_w   = (const float*)d_in[12];
  const float* gl_b   = (const float*)d_in[13];

  char* p = (char*)d_ws;
  auto take = [&](size_t bytes)->char* {
    char* r = p; p += (bytes + 255) & ~(size_t)255; return r;
  };
  __bf16* left   = (__bf16*)take((size_t)NM*CZ*2);   // 64 MB
  __bf16* Wcat   = (__bf16*)take((size_t)640*128*2);
  __bf16* Wout   = (__bf16*)take((size_t)128*128*2);
  __bf16* leftp  = (__bf16*)take((size_t)CI*NM*2);   // 64 MB, [c][i*512+k]
  __bf16* rightp = (__bf16*)take((size_t)CI*NM*2);   // 64 MB, [c][j*512+k]
  float*  glg    = (float*) take((size_t)NM*CI*4);   // 128 MB, [m][c]
  float*  opre   = (float*) take((size_t)CI*NM*4);   // 128 MB, [c][i*512+j]
  float*  meanA  = (float*) take((size_t)NM*4);
  float*  rstdA  = (float*) take((size_t)NM*4);

  k0_cast_weights<<<384, 256, 0, stream>>>(proj_w, gate_w, gl_w, outp_w, Wcat, Wout);
  k1_ln_cast<<<NM/8, 256, 0, stream>>>(act, ln_w, ln_b, left);
  k2_proj_gate<<<dim3(NM/64, 4), 256, 0, stream>>>(left, Wcat, mask, proj_b, gate_b, gl_b,
                                                   leftp, rightp, glg);
  k3_einsum<<<dim3(N_/64, N_/64, CI), 128, 0, stream>>>(leftp, rightp, opre);
  k4a_stats<<<NM/256, 256, 0, stream>>>(opre, meanA, rstdA);
  k4b_outproj<<<NM/64, 256, 0, stream>>>(opre, meanA, rstdA, cn_w, cn_b, Wout, outp_b,
                                         glg, (float*)d_out);
}
